// DiSA_43885975831119
// MI455X (gfx1250) — matrix operand order repaired and checked by reading, not compiled, not run on hardware
//
#include <hip/hip_runtime.h>
#include <hip/hip_bf16.h>

// ---------------------------------------------------------------------------
// DiSA forward for MI455X (gfx1250, wave32).
// GEMMs via v_wmma_f32_16x16x32_bf16 with 2-term bf16 error-split (~fp32 acc).
// Attention (per-feature masked softmax) fused with online softmax (VALU).
// All padded dims = 320 so WMMA waves run with EXEC all-1s, no guards.
// ---------------------------------------------------------------------------

typedef __attribute__((ext_vector_type(16))) __bf16 v16bf;
typedef __attribute__((ext_vector_type(8)))  __bf16 v8bf;
typedef __attribute__((ext_vector_type(8)))  float  v8f;

#define KP 320                 // padded K (and N) dim
#define MROWS 1024             // B*S
#define NTILE 20               // 320/16 output-column tiles
#define MTILE 64               // 1024/16 output-row tiles
#define GEMM_WAVES (MTILE*NTILE)   // 1280 tiles, one wave each
#define GEMM_BLOCKS (GEMM_WAVES/4) // 4 waves (128 thr) per block

#define WMMA_BF16(A,B,C) \
  __builtin_amdgcn_wmma_f32_16x16x32_bf16(false,(A),false,(B),(short)0,(C),false,false)

// ----- fragment loaders (contiguous k order, same for A and B, wave32) -----------------------
// A (16x32, bf16): lane m (m<16) holds row m, K = k0+0..15;
//                  lane m+16 holds row m, K = k0+16..31.
__device__ __forceinline__ v16bf load_a_frag(const __bf16* __restrict__ A,
                                             int row0, int k0, int lane) {
  const int half = lane >> 4, lm = lane & 15;
  const __bf16* p = A + (size_t)(row0 + lm) * KP + k0 + half * 16;
  v8bf x0 = *(const v8bf*)(p);
  v8bf x1 = *(const v8bf*)(p + 8);
  v16bf r;
#pragma unroll
  for (int e = 0; e < 8; ++e) { r[e] = x0[e]; r[e + 8] = x1[e]; }
  return r;
}

// B (32x16, bf16), B[k][n] = W[n][k] (W row-major [N,KP]):
// lane n (n<16) holds col n, K = k0+0..15; lane n+16 holds col n, K = k0+16..31.
__device__ __forceinline__ v16bf load_b_frag(const __bf16* __restrict__ W,
                                             int col0, int k0, int lane) {
  const int half = lane >> 4, lm = lane & 15;
  return *(const v16bf*)(W + (size_t)(col0 + lm) * KP + k0 + half * 16);
}

// ----- pack fp32 -> (hi, lo) bf16 split, zero-padded -----------------------
__global__ void pack_split_kernel(const float* __restrict__ src, int srows, int scols,
                                  int prows, int pcols,
                                  __bf16* __restrict__ hi, __bf16* __restrict__ lo) {
  int idx = blockIdx.x * blockDim.x + threadIdx.x;
  if (idx >= prows * pcols) return;
  int r = idx / pcols, c = idx - r * pcols;
  float v = (r < srows && c < scols) ? src[(size_t)r * scols + c] : 0.0f;
  __bf16 h = (__bf16)v;
  hi[idx] = h;
  lo[idx] = (__bf16)(v - (float)h);
}

// ----- GEMM1: rep = elu(X @ W_fc^T + b_fc); also re-split rep to bf16 ------
__global__ __launch_bounds__(128) void gemm1_repmap(
    const __bf16* __restrict__ Xhi, const __bf16* __restrict__ Xlo,
    const __bf16* __restrict__ Whi, const __bf16* __restrict__ Wlo,
    const float* __restrict__ bfc,
    float* __restrict__ rep, __bf16* __restrict__ rephi, __bf16* __restrict__ replo) {
  const int wave = (blockIdx.x * blockDim.x + threadIdx.x) >> 5;
  const int lane = threadIdx.x & 31;
  const int tm = wave / NTILE, tn = wave - tm * NTILE;
  const int row0 = tm * 16, col0 = tn * 16;
  v8f c = {};
  for (int k0 = 0; k0 < KP; k0 += 32) {
    v16bf ah = load_a_frag(Xhi, row0, k0, lane);
    v16bf al = load_a_frag(Xlo, row0, k0, lane);
    v16bf bh = load_b_frag(Whi, col0, k0, lane);
    v16bf bl = load_b_frag(Wlo, col0, k0, lane);
    c = WMMA_BF16(ah, bh, c);
    c = WMMA_BF16(ah, bl, c);
    c = WMMA_BF16(al, bh, c);
  }
  const int lm = lane & 15, half = lane >> 4;
  const int col = col0 + lm;
  const float bias = (col < 300) ? bfc[col] : 0.0f;
#pragma unroll
  for (int v = 0; v < 8; ++v) {
    int row = row0 + v + half * 8;
    float x = c[v] + bias;
    float y = (x > 0.0f) ? x : (__expf(x) - 1.0f);   // elu, alpha=1
    if (col >= 300) y = 0.0f;                        // keep padding exactly 0
    size_t o = (size_t)row * KP + col;
    rep[o] = y;
    __bf16 h = (__bf16)y;
    rephi[o] = h;
    replo[o] = (__bf16)(y - (float)h);
  }
}

// ----- GEMM2 (dual): dep = rep @ W1^T, head = rep @ W2^T -------------------
__global__ __launch_bounds__(128) void gemm2_dephead(
    const __bf16* __restrict__ Ahi, const __bf16* __restrict__ Alo,
    const __bf16* __restrict__ W1hi, const __bf16* __restrict__ W1lo,
    const __bf16* __restrict__ W2hi, const __bf16* __restrict__ W2lo,
    float* __restrict__ dep, float* __restrict__ head) {
  const int wave = (blockIdx.x * blockDim.x + threadIdx.x) >> 5;
  const int lane = threadIdx.x & 31;
  const int tm = wave / NTILE, tn = wave - tm * NTILE;
  const int row0 = tm * 16, col0 = tn * 16;
  v8f c1 = {}, c2 = {};
  for (int k0 = 0; k0 < KP; k0 += 32) {
    v16bf ah = load_a_frag(Ahi, row0, k0, lane);
    v16bf al = load_a_frag(Alo, row0, k0, lane);
    v16bf b1h = load_b_frag(W1hi, col0, k0, lane);
    v16bf b1l = load_b_frag(W1lo, col0, k0, lane);
    v16bf b2h = load_b_frag(W2hi, col0, k0, lane);
    v16bf b2l = load_b_frag(W2lo, col0, k0, lane);
    c1 = WMMA_BF16(ah, b1h, c1);
    c1 = WMMA_BF16(ah, b1l, c1);
    c1 = WMMA_BF16(al, b1h, c1);
    c2 = WMMA_BF16(ah, b2h, c2);
    c2 = WMMA_BF16(ah, b2l, c2);
    c2 = WMMA_BF16(al, b2h, c2);
  }
  const int lm = lane & 15, half = lane >> 4;
  const int col = col0 + lm;
#pragma unroll
  for (int v = 0; v < 8; ++v) {
    int row = row0 + v + half * 8;
    size_t o = (size_t)row * KP + col;
    dep[o]  = c1[v];   // padded cols are exactly 0 (zero-padded operands)
    head[o] = c2[v];
  }
}

// ----- fused per-feature masked softmax attention (online) -----------------
// block = (b,i) pair (grid 1024), thread = feature d (blockDim 320; d<300 live).
// Faithful to ref: the max includes the 0 entries of masked logits -> m0 = 0.
__global__ __launch_bounds__(320) void attention_kernel(
    const float* __restrict__ dep, const float* __restrict__ head,
    const float* __restrict__ rep, const float* __restrict__ bias1,
    const int* __restrict__ rep_mask,
    float* __restrict__ attn, __bf16* __restrict__ athi, __bf16* __restrict__ atlo) {
  const int bi = blockIdx.x;          // b*256 + i
  const int b = bi >> 8, i = bi & 255;
  const int d = threadIdx.x;          // 0..319
  const bool live = (d < 300);
  const float* depb = dep + (size_t)b * 256 * KP;
  const float* repb = rep + (size_t)b * 256 * KP;
  const float hd = live ? head[(size_t)bi * KP + d] : 0.0f;
  const float bb = live ? bias1[d] : 0.0f;
  const float invC = 0.2f;            // 1/C, C = 5
  float m = 0.0f, s = 0.0f, acc = 0.0f;   // m=0 models the masked-out zeros
  for (int j = i + 1; j < 256; ++j) {
    if (rep_mask[b * 256 + j] == 0) continue;   // uniform across block
    float l = 5.0f * tanhf((depb[(size_t)j * KP + d] + hd + bb) * invC);
    float mn = fmaxf(m, l);
    float sc = __expf(m - mn);
    float e  = __expf(l - mn);
    s   = s * sc + e;
    acc = acc * sc + e * repb[(size_t)j * KP + d];
    m = mn;
  }
  float denom = ((s == 0.0f) ? 1.0f : s) + 1e-20f;
  float r = live ? (acc / denom) : 0.0f;
  size_t o = (size_t)bi * KP + d;
  attn[o] = r;
  __bf16 h = (__bf16)r;
  athi[o] = h;
  atlo[o] = (__bf16)(r - (float)h);
}

// ----- GEMM3 (dual) + gate epilogue: out = mask*(g*rep + (1-g)*attn) -------
__global__ __launch_bounds__(128) void gemm3_gate_out(
    const __bf16* __restrict__ Rhi, const __bf16* __restrict__ Rlo,
    const __bf16* __restrict__ Thi, const __bf16* __restrict__ Tlo,
    const __bf16* __restrict__ F1hi, const __bf16* __restrict__ F1lo,
    const __bf16* __restrict__ F2hi, const __bf16* __restrict__ F2lo,
    const float* __restrict__ bf, const float* __restrict__ rep,
    const float* __restrict__ attn, const int* __restrict__ rep_mask,
    float* __restrict__ out) {
  const int wave = (blockIdx.x * blockDim.x + threadIdx.x) >> 5;
  const int lane = threadIdx.x & 31;
  const int tm = wave / NTILE, tn = wave - tm * NTILE;
  const int row0 = tm * 16, col0 = tn * 16;
  v8f c1 = {}, c2 = {};
  for (int k0 = 0; k0 < KP; k0 += 32) {
    v16bf ah  = load_a_frag(Rhi, row0, k0, lane);
    v16bf al  = load_a_frag(Rlo, row0, k0, lane);
    v16bf th  = load_a_frag(Thi, row0, k0, lane);
    v16bf tl  = load_a_frag(Tlo, row0, k0, lane);
    v16bf b1h = load_b_frag(F1hi, col0, k0, lane);
    v16bf b1l = load_b_frag(F1lo, col0, k0, lane);
    v16bf b2h = load_b_frag(F2hi, col0, k0, lane);
    v16bf b2l = load_b_frag(F2lo, col0, k0, lane);
    c1 = WMMA_BF16(ah, b1h, c1);
    c1 = WMMA_BF16(ah, b1l, c1);
    c1 = WMMA_BF16(al, b1h, c1);
    c2 = WMMA_BF16(th, b2h, c2);
    c2 = WMMA_BF16(th, b2l, c2);
    c2 = WMMA_BF16(tl, b2h, c2);
  }
  const int lm = lane & 15, half = lane >> 4;
  const int col = col0 + lm;
  if (col < 300) {
    const float bias = bf[col];
#pragma unroll
    for (int v = 0; v < 8; ++v) {
      int row = row0 + v + half * 8;
      float g = 1.0f / (1.0f + __expf(-(c1[v] + c2[v] + bias)));
      size_t o = (size_t)row * KP + col;
      float rm = rep[o], ar = attn[o];
      float y = g * rm + (1.0f - g) * ar;
      out[(size_t)row * 300 + col] = y * (float)rep_mask[row];
    }
  }
}

// ---------------------------------------------------------------------------
extern "C" void kernel_launch(void* const* d_in, const int* in_sizes, int n_in,
                              void* d_out, int out_size, void* d_ws, size_t ws_size,
                              hipStream_t stream) {
  const float* X   = (const float*)d_in[0];   // [4,256,300]
  const int*   rm  = (const int*)  d_in[1];   // [4,256]
  const float* Wfc = (const float*)d_in[2];   // [300,300]
  const float* bfc = (const float*)d_in[3];
  const float* W1  = (const float*)d_in[4];
  const float* W2  = (const float*)d_in[5];
  const float* b1  = (const float*)d_in[6];
  const float* Wf1 = (const float*)d_in[7];
  const float* Wf2 = (const float*)d_in[8];
  const float* bf  = (const float*)d_in[9];
  float* out = (float*)d_out;
  (void)in_sizes; (void)n_in; (void)out_size; (void)ws_size;

  char* base = (char*)d_ws;
  size_t off = 0;
  auto carve = [&](size_t bytes) -> void* {
    void* p = base + off;
    off = (off + bytes + 255) & ~(size_t)255;
    return p;
  };
  const size_t actB = (size_t)MROWS * KP * sizeof(__bf16);   // 640 KB
  const size_t wB   = (size_t)KP * KP * sizeof(__bf16);      // 200 KB
  const size_t actF = (size_t)MROWS * KP * sizeof(float);    // 1.25 MB

  __bf16 *Xhi = (__bf16*)carve(actB), *Xlo = (__bf16*)carve(actB);
  __bf16 *Wfch = (__bf16*)carve(wB), *Wfcl = (__bf16*)carve(wB);
  __bf16 *W1h  = (__bf16*)carve(wB), *W1l  = (__bf16*)carve(wB);
  __bf16 *W2h  = (__bf16*)carve(wB), *W2l  = (__bf16*)carve(wB);
  __bf16 *Wf1h = (__bf16*)carve(wB), *Wf1l = (__bf16*)carve(wB);
  __bf16 *Wf2h = (__bf16*)carve(wB), *Wf2l = (__bf16*)carve(wB);
  __bf16 *Rhi = (__bf16*)carve(actB), *Rlo = (__bf16*)carve(actB);
  __bf16 *Thi = (__bf16*)carve(actB), *Tlo = (__bf16*)carve(actB);
  float *rep  = (float*)carve(actF);
  float *dep  = (float*)carve(actF);
  float *head = (float*)carve(actF);
  float *attn = (float*)carve(actF);   // total ~10.7 MB

  // 1) pack + bf16 error-split (zero-padded to 320)
  {
    int nX = MROWS * KP;
    pack_split_kernel<<<(nX + 255) / 256, 256, 0, stream>>>(X, MROWS, 300, MROWS, KP, Xhi, Xlo);
    int nW = KP * KP;
    int gw = (nW + 255) / 256;
    pack_split_kernel<<<gw, 256, 0, stream>>>(Wfc, 300, 300, KP, KP, Wfch, Wfcl);
    pack_split_kernel<<<gw, 256, 0, stream>>>(W1,  300, 300, KP, KP, W1h,  W1l);
    pack_split_kernel<<<gw, 256, 0, stream>>>(W2,  300, 300, KP, KP, W2h,  W2l);
    pack_split_kernel<<<gw, 256, 0, stream>>>(Wf1, 300, 300, KP, KP, Wf1h, Wf1l);
    pack_split_kernel<<<gw, 256, 0, stream>>>(Wf2, 300, 300, KP, KP, Wf2h, Wf2l);
  }
  // 2) rep_map = elu(X @ Wfc^T + b_fc)
  gemm1_repmap<<<GEMM_BLOCKS, 128, 0, stream>>>(Xhi, Xlo, Wfch, Wfcl, bfc, rep, Rhi, Rlo);
  // 3) dep / head
  gemm2_dephead<<<GEMM_BLOCKS, 128, 0, stream>>>(Rhi, Rlo, W1h, W1l, W2h, W2l, dep, head);
  // 4) fused per-feature masked softmax attention
  attention_kernel<<<MROWS, KP, 0, stream>>>(dep, head, rep, b1, rm, attn, Thi, Tlo);
  // 5) gate GEMMs + fused sigmoid/mix/mask epilogue
  gemm3_gate_out<<<GEMM_BLOCKS, 128, 0, stream>>>(Rhi, Rlo, Thi, Tlo, Wf1h, Wf1l,
                                                  Wf2h, Wf2l, bf, rep, attn, rm, out);
}
